// Attention_75943611728690
// MI455X (gfx1250) — matrix operand order repaired and checked by reading, not compiled, not run on hardware
//
#include <hip/hip_runtime.h>
#include <hip/hip_bf16.h>

typedef __bf16 bf16_t;
typedef __bf16 v16bf __attribute__((ext_vector_type(16)));
typedef __bf16 v8bf  __attribute__((ext_vector_type(8)));
typedef __bf16 v4bf  __attribute__((ext_vector_type(4)));
typedef float  v8f   __attribute__((ext_vector_type(8)));
typedef float  v4f   __attribute__((ext_vector_type(4)));

// pointer types for the CDNA5 async-to-LDS builtin:
// param0 = int4 in addrspace(1) (global), param1 = int4 in addrspace(3) (LDS)
typedef int v4i __attribute__((vector_size(16)));
typedef __attribute__((address_space(1))) v4i* gv4i_ptr;
typedef __attribute__((address_space(3))) v4i* lv4i_ptr;

#define DIM  4096
#define NH   32
#define HD   128
#define BSZ  2
#define SEQ  2048
#define MTOT (BSZ * SEQ)

// LDS row padding: +8 bf16 (16B) keeps 16B alignment but skews rows by
// 4 banks, de-conflicting column-wise fragment reads.
#define SSTRIDE (SEQ + 8)
#define VSTRIDE (HD + 8)

static __device__ __forceinline__ v16bf join16(v8bf lo, v8bf hi) {
    v16bf r;
#pragma unroll
    for (int i = 0; i < 8; ++i) { r[i] = lo[i]; r[8 + i] = hi[i]; }
    return r;
}

// ---------------------------------------------------------------------------
// fp32 -> bf16 conversion (vectorized x4)
// ---------------------------------------------------------------------------
__global__ void cvt_f32_bf16(const float* __restrict__ src,
                             bf16_t* __restrict__ dst, int n) {
    int i = (blockIdx.x * blockDim.x + threadIdx.x) * 4;
    if (i + 3 < n) {
        v4f v = *reinterpret_cast<const v4f*>(src + i);
        v4bf o;
        o[0] = (bf16_t)v[0]; o[1] = (bf16_t)v[1];
        o[2] = (bf16_t)v[2]; o[3] = (bf16_t)v[3];
        *reinterpret_cast<v4bf*>(dst + i) = o;
    }
}

// ---------------------------------------------------------------------------
// Fused QKV projection + RoPE epilogue.
// Q = rope(x@Wq^T), K = rope(x@Wk^T), V = x@Wv^T  (bf16 in, bf16 out)
// A fragment loaded once per k-step and reused for 12 WMMAs; the three
// B fragments per n-subtile are loaded together so the compiler can clause
// 6 b128 loads ahead of 3 back-to-back WMMAs.
// ---------------------------------------------------------------------------
__global__ __launch_bounds__(256) void gemm_qkv_rope(
    const bf16_t* __restrict__ A,
    const bf16_t* __restrict__ Wq, const bf16_t* __restrict__ Wk,
    const bf16_t* __restrict__ Wv,
    bf16_t* __restrict__ Qo, bf16_t* __restrict__ Ko, bf16_t* __restrict__ Vo,
    const float* __restrict__ fcos, const float* __restrict__ fsin) {
    const int K = DIM, N = DIM;
    const int lane  = threadIdx.x & 31;
    const int wave  = threadIdx.x >> 5;
    const int mBase = blockIdx.y * 64 + (wave & 3) * 16;
    const int nBase = blockIdx.x * 128 + (wave >> 2) * 64;
    const int half  = lane >> 4;
    const int l16   = lane & 15;

    v8f aq[4] = {}, ak[4] = {}, av[4] = {};

    const size_t aRow = (size_t)(mBase + l16) * K;
    for (int kk = 0; kk < K; kk += 32) {
        const bf16_t* ap = A + aRow + kk + half * 16;
        v16bf afrag = join16(*(const v8bf*)ap, *(const v8bf*)(ap + 8));
        __builtin_prefetch(ap + 32, 0, 1);
#pragma unroll
        for (int t = 0; t < 4; ++t) {
            const size_t bro = (size_t)(nBase + t * 16 + l16) * K + kk + half * 16;
            const bf16_t* bq = Wq + bro;
            const bf16_t* bk = Wk + bro;
            const bf16_t* bv = Wv + bro;
            v16bf fq = join16(*(const v8bf*)bq, *(const v8bf*)(bq + 8));
            v16bf fk = join16(*(const v8bf*)bk, *(const v8bf*)(bk + 8));
            v16bf fv = join16(*(const v8bf*)bv, *(const v8bf*)(bv + 8));
            aq[t] = __builtin_amdgcn_wmma_f32_16x16x32_bf16(
                false, afrag, false, fq, (short)0, aq[t], false, false);
            ak[t] = __builtin_amdgcn_wmma_f32_16x16x32_bf16(
                false, afrag, false, fk, (short)0, ak[t], false, false);
            av[t] = __builtin_amdgcn_wmma_f32_16x16x32_bf16(
                false, afrag, false, fv, (short)0, av[t], false, false);
        }
    }

    const bool even = (l16 & 1) == 0;
#pragma unroll
    for (int t = 0; t < 4; ++t) {
#pragma unroll
        for (int v = 0; v < 8; ++v) {
            const int row = mBase + v + half * 8;      // token index
            const int col = nBase + t * 16 + l16;      // feature index
            const int s   = row & (SEQ - 1);
            const int i   = (col & (HD - 1)) >> 1;
            const float c  = fcos[s * 64 + i];
            const float sn = fsin[s * 64 + i];
            const size_t off = (size_t)row * N + col;

            // Q with rotary (pair exchange via shfl_xor lane^1)
            {
                float own = aq[t][v];
                float oth = __shfl_xor(own, 1, 32);
                float t0 = even ? own : oth;
                float t1 = even ? oth : own;
                float r  = even ? (t0 * c - t1 * sn) : (t0 * sn + t1 * c);
                Qo[off] = (bf16_t)r;
            }
            // K with rotary
            {
                float own = ak[t][v];
                float oth = __shfl_xor(own, 1, 32);
                float t0 = even ? own : oth;
                float t1 = even ? oth : own;
                float r  = even ? (t0 * c - t1 * sn) : (t0 * sn + t1 * c);
                Ko[off] = (bf16_t)r;
            }
            // V plain
            Vo[off] = (bf16_t)av[t][v];
        }
    }
}

// ---------------------------------------------------------------------------
// C[M,N] = A[M,K] * B[N,K]^T  (generic WMMA GEMM, used for output projection)
// ---------------------------------------------------------------------------
template <typename OutT>
__global__ __launch_bounds__(256) void gemm_bt(const bf16_t* __restrict__ A,
                                               const bf16_t* __restrict__ B,
                                               OutT* __restrict__ C,
                                               int M, int N, int K) {
    const int lane  = threadIdx.x & 31;
    const int wave  = threadIdx.x >> 5;
    const int mBase = blockIdx.y * 64 + (wave & 3) * 16;
    const int nBase = blockIdx.x * 128 + (wave >> 2) * 64;
    const int half  = lane >> 4;
    const int l16   = lane & 15;

    v8f acc[4] = {};

    const size_t aRow = (size_t)(mBase + l16) * K;
    for (int kk = 0; kk < K; kk += 32) {
        const bf16_t* ap = A + aRow + kk + half * 16;
        v16bf afrag = join16(*(const v8bf*)ap, *(const v8bf*)(ap + 8));
        __builtin_prefetch(ap + 32, 0, 1);
#pragma unroll
        for (int t = 0; t < 4; ++t) {
            const bf16_t* bp = B + (size_t)(nBase + t * 16 + l16) * K + kk + half * 16;
            v16bf bfrag = join16(*(const v8bf*)bp, *(const v8bf*)(bp + 8));
            acc[t] = __builtin_amdgcn_wmma_f32_16x16x32_bf16(
                false, afrag, false, bfrag, (short)0, acc[t], false, false);
        }
    }

#pragma unroll
    for (int t = 0; t < 4; ++t) {
#pragma unroll
        for (int v = 0; v < 8; ++v) {
            int row = mBase + v + half * 8;
            int col = nBase + t * 16 + l16;
            C[(size_t)row * N + col] = (OutT)acc[t][v];
        }
    }
}

// ---------------------------------------------------------------------------
// Causal attention, one block per (32-query block, head, batch).
// Pass 1: scores (32 x SEQ) -> LDS as bf16 via WMMA (causal mask applied).
// Pass 2: row softmax (8 threads/row, shuffle reductions); unnormalized
//         probs kept in LDS, 1/sum in sInv.
// Pass 3: O = P*V via WMMA; 32-key V chunks staged into LDS, preferring the
//         CDNA5 async-to-LDS path (ASYNCcnt) when the toolchain exposes it.
// ---------------------------------------------------------------------------
__global__ __launch_bounds__(256, 1) void attn_kernel(
    const bf16_t* __restrict__ Q, const bf16_t* __restrict__ Kt,
    const bf16_t* __restrict__ V, bf16_t* __restrict__ O) {
    __shared__ bf16_t sS[32][SSTRIDE];   // ~128.5 KB score/prob buffer (bf16)
    __shared__ bf16_t sV[32][VSTRIDE];   // ~8.5 KB V staging tile
    __shared__ float  sInv[32];

    const int h    = blockIdx.y;
    const int b    = blockIdx.z;
    const int tid  = threadIdx.x;
    const int lane = tid & 31;
    const int wave = tid >> 5;
    const int half = lane >> 4;
    const int l16  = lane & 15;
    const int qb32 = blockIdx.x * 32;

    const size_t batchOff = (size_t)b * SEQ * DIM;
    const size_t headOff  = (size_t)h * HD;
    const float  scale    = 0.08838834764831845f;  // 1/sqrt(128)

    // ---- Pass 1: scores -> LDS (bf16) ----
    {
        const int qt2 = wave & 1;            // which 16-row subtile
        v16bf qf[4];
        const bf16_t* qrow =
            Q + batchOff + (size_t)(qb32 + qt2 * 16 + l16) * DIM + headOff;
#pragma unroll
        for (int c = 0; c < 4; ++c) {
            const bf16_t* p = qrow + c * 32 + half * 16;
            qf[c] = join16(*(const v8bf*)p, *(const v8bf*)(p + 8));
        }
        for (int j = 0; j < 32; ++j) {
            const int kt = (wave >> 1) + 4 * j;
            v8f acc = {};
            const bf16_t* krow =
                Kt + batchOff + (size_t)(kt * 16 + l16) * DIM + headOff;
#pragma unroll
            for (int c = 0; c < 4; ++c) {
                const bf16_t* p = krow + c * 32 + half * 16;
                v16bf kf = join16(*(const v8bf*)p, *(const v8bf*)(p + 8));
                acc = __builtin_amdgcn_wmma_f32_16x16x32_bf16(
                    false, qf[c], false, kf, (short)0, acc, false, false);
            }
#pragma unroll
            for (int v = 0; v < 8; ++v) {
                const int m    = qt2 * 16 + v + half * 8;   // local q row 0..31
                const int kidx = kt * 16 + l16;
                float val = acc[v] * scale;
                if (kidx > qb32 + m) val = -1.0e9f;         // causal mask
                sS[m][kidx] = (bf16_t)val;
            }
        }
    }
    __syncthreads();

    // ---- Pass 2: softmax (8 threads per row) ----
    {
        const int r = tid >> 3, sub = tid & 7;
        float mx = -3.4e38f;
        for (int i = sub; i < SEQ; i += 8) mx = fmaxf(mx, (float)sS[r][i]);
#pragma unroll
        for (int m = 1; m < 8; m <<= 1) mx = fmaxf(mx, __shfl_xor(mx, m, 32));
        float sum = 0.0f;
        for (int i = sub; i < SEQ; i += 8) {
            float e = __expf((float)sS[r][i] - mx);
            sS[r][i] = (bf16_t)e;
            sum += e;
        }
#pragma unroll
        for (int m = 1; m < 8; m <<= 1) sum += __shfl_xor(sum, m, 32);
        if (sub == 0) sInv[r] = 1.0f / sum;
    }
    __syncthreads();

    // ---- Pass 3: O = P * V ----
    const int qrow0  = (wave >> 2) * 16;    // 0 or 16
    const int dsBase = (wave & 3) * 32;     // 0,32,64,96
    v8f accO[2] = {};
    for (int kc = 0; kc < SEQ / 32; ++kc) {
        // stage 32x128 V chunk into LDS
        {
            const int row = tid >> 3;           // 0..31
            const int cg  = (tid & 7) * 16;     // 0,16,...,112
            const bf16_t* vp =
                V + batchOff + (size_t)(kc * 32 + row) * DIM + headOff + cg;
#if __has_builtin(__builtin_amdgcn_global_load_async_to_lds_b128)
            // CDNA5 async DMA: memory -> LDS without touching VGPRs
            __builtin_amdgcn_global_load_async_to_lds_b128(
                (gv4i_ptr)(vp),
                (lv4i_ptr)(&sV[row][cg]), 0, 0);
            __builtin_amdgcn_global_load_async_to_lds_b128(
                (gv4i_ptr)(vp + 8),
                (lv4i_ptr)(&sV[row][cg + 8]), 0, 0);
#if __has_builtin(__builtin_amdgcn_s_wait_asynccnt)
            __builtin_amdgcn_s_wait_asynccnt(0);
#else
            asm volatile("s_wait_asynccnt 0x0" ::: "memory");
#endif
#else
            *(v8bf*)&sV[row][cg]     = *(const v8bf*)vp;
            *(v8bf*)&sV[row][cg + 8] = *(const v8bf*)(vp + 8);
#endif
        }
        __syncthreads();

        // P fragment: A-layout, row = qrow0 + l16 (direct bf16 copy from LDS)
        v16bf pf;
#pragma unroll
        for (int i = 0; i < 16; ++i) {
            const int k = half * 16 + i;
            pf[i] = sS[qrow0 + l16][kc * 32 + k];
        }
        // V fragments: B-layout, cols dsBase+l16 and dsBase+16+l16
        v16bf vf0, vf1;
#pragma unroll
        for (int i = 0; i < 16; ++i) {
            const int k = half * 16 + i;
            vf0[i] = sV[k][dsBase + l16];
            vf1[i] = sV[k][dsBase + 16 + l16];
        }
        accO[0] = __builtin_amdgcn_wmma_f32_16x16x32_bf16(
            false, pf, false, vf0, (short)0, accO[0], false, false);
        accO[1] = __builtin_amdgcn_wmma_f32_16x16x32_bf16(
            false, pf, false, vf1, (short)0, accO[1], false, false);
        __syncthreads();
    }

#pragma unroll
    for (int t = 0; t < 2; ++t) {
#pragma unroll
        for (int v = 0; v < 8; ++v) {
            const int m   = qrow0 + v + half * 8;       // local q row
            const int col = dsBase + t * 16 + l16;      // head_dim column
            const float val = accO[t][v] * sInv[m];
            O[batchOff + (size_t)(qb32 + m) * DIM + headOff + col] = (bf16_t)val;
        }
    }
}

// ---------------------------------------------------------------------------
// Host-side launcher
// Inputs: 0=x 1=start_pos 2=freqs_cos 3=freqs_sin 4=mask 5=wq 6=wk 7=wv 8=wo
// ---------------------------------------------------------------------------
extern "C" void kernel_launch(void* const* d_in, const int* in_sizes, int n_in,
                              void* d_out, int out_size, void* d_ws, size_t ws_size,
                              hipStream_t stream) {
    const float* x    = (const float*)d_in[0];
    const float* fcos = (const float*)d_in[2];
    const float* fsin = (const float*)d_in[3];
    const float* wq   = (const float*)d_in[5];
    const float* wk   = (const float*)d_in[6];
    const float* wv   = (const float*)d_in[7];
    const float* wo   = (const float*)d_in[8];
    float* out = (float*)d_out;

    char* ws = (char*)d_ws;
    const size_t MB = 1024ull * 1024ull;
    bf16_t* xb  = (bf16_t*)(ws + 0 * MB);
    bf16_t* wqb = (bf16_t*)(ws + 32 * MB);
    bf16_t* wkb = (bf16_t*)(ws + 64 * MB);
    bf16_t* wvb = (bf16_t*)(ws + 96 * MB);
    bf16_t* wob = (bf16_t*)(ws + 128 * MB);
    bf16_t* Qb  = (bf16_t*)(ws + 160 * MB);
    bf16_t* Kb  = (bf16_t*)(ws + 192 * MB);
    bf16_t* Vb  = (bf16_t*)(ws + 224 * MB);
    bf16_t* Ob  = (bf16_t*)(ws + 256 * MB);

    // 1) convert activations + weights to bf16
    const int nTok = MTOT * DIM;
    const int nW   = DIM * DIM;
    cvt_f32_bf16<<<nTok / 1024, 256, 0, stream>>>(x,  xb,  nTok);
    cvt_f32_bf16<<<nW   / 1024, 256, 0, stream>>>(wq, wqb, nW);
    cvt_f32_bf16<<<nW   / 1024, 256, 0, stream>>>(wk, wkb, nW);
    cvt_f32_bf16<<<nW   / 1024, 256, 0, stream>>>(wv, wvb, nW);
    cvt_f32_bf16<<<nW   / 1024, 256, 0, stream>>>(wo, wob, nW);

    // 2) fused Q/K/V projections + RoPE (x read once, A fragment reused 12x)
    dim3 gg(DIM / 128, MTOT / 64);
    gemm_qkv_rope<<<gg, 256, 0, stream>>>(xb, wqb, wkb, wvb,
                                          Qb, Kb, Vb, fcos, fsin);

    // 3) causal attention (32 query rows per block)
    dim3 ag(SEQ / 32, NH, BSZ);
    attn_kernel<<<ag, 256, 0, stream>>>(Qb, Kb, Vb, Ob);

    // 4) output projection -> fp32 result
    gemm_bt<float><<<gg, 256, 0, stream>>>(Ob, wob, out, MTOT, DIM, DIM);
}
